// ValueNetOld_6270652252640
// MI455X (gfx1250) — compile-verified, matrix operand order checked by reading, not run on hardware
//
#include <hip/hip_runtime.h>
#include <hip/hip_bf16.h>

// ---------------- problem constants ----------------
#define NN   20000      // nodes
#define NE   256000     // edges
#define NESL 276000     // edges + self loops
#define NBATCH 256
#define HD1 64
#define HD2 128
#define NHEADS 8
#define HV 512          // HEADS*HD1
#define KXG 160         // 131 padded to mult of 32
#define KXS 544         // 523 padded to mult of 32
#define MT  1250        // NN/16 tiles
#define MITER 5         // M-tiles per block (amortize LDS-staged B panel)
#define MBLK (MT/MITER) // 250

#define CDIV(a,b) (((a)+(b)-1)/(b))

typedef __attribute__((ext_vector_type(16))) _Float16 v16h;
typedef __attribute__((ext_vector_type(8)))  _Float16 v8h;
typedef __attribute__((ext_vector_type(8)))  float    v8f;

// ---------------- small helpers ----------------
__device__ inline void atomicMaxFloat(float* a, float v) {
  // order-preserving encoding: works with buffer initialized to -inf
  if (v >= 0.0f) atomicMax((int*)a, __float_as_int(v));
  else           atomicMin((unsigned int*)a, (unsigned int)__float_as_int(v));
}

__global__ void k_fillf(float* p, float v, int n) {
  int i = blockIdx.x * blockDim.x + threadIdx.x;
  if (i < n) p[i] = v;
}
__global__ void k_copyf(const float* a, float* b, int n) {
  int i = blockIdx.x * blockDim.x + threadIdx.x;
  if (i < n) b[i] = a[i];
}
__global__ void k_cvt16(const float* in, _Float16* out, int n) {
  int i = blockIdx.x * blockDim.x + threadIdx.x;
  if (i < n) out[i] = (_Float16)in[i];
}
// convert [rows,incols] f32 -> [rows,outcols] f16, zero padded
__global__ void k_cvt16_pad(const float* in, _Float16* out, int rows, int incols, int outcols) {
  int i = blockIdx.x * blockDim.x + threadIdx.x;
  if (i >= rows * outcols) return;
  int r = i / outcols, c = i % outcols;
  out[i] = (c < incols) ? (_Float16)in[r * incols + c] : (_Float16)0.0f;
}
__global__ void k_addf(const float* a, const float* b, float* o, int n) {
  int i = blockIdx.x * blockDim.x + threadIdx.x;
  if (i < n) o[i] = a[i] + b[i];
}

// ---------------- WMMA GEMM: C[M,Nc] = act(A[M,K] @ W[Nc,K]^T + bias) ----------------
// grid.x = MT/MITER, grid.y = Nc/64, block = 128 (4 waves, one 16x16 N-tile each).
// USE_LDS: block async-copies its 64-row B panel to LDS once
// (global_load_async_to_lds_b128 / ASYNCcnt) and reuses it for MITER M-tiles.
// K and USE_LDS are compile-time so the inner loop is straight-line WMMA.
template<int K, bool USE_LDS>
__global__ __launch_bounds__(128)
void k_wmma_gemm(const _Float16* __restrict__ A, const _Float16* __restrict__ Bw,
                 const float* __restrict__ bias, float* __restrict__ C,
                 _Float16* __restrict__ C16, int Ncols, int act) {
  const int lane = threadIdx.x & 31;
  const int wave = threadIdx.x >> 5;
  const int row  = lane & 15;      // M-row for A frag, column for B frag
  const int grp  = lane >> 4;      // lane group (K interleave)
  const int tn   = blockIdx.y * 4 + wave;
  const int ncol = tn * 16 + row;

  __shared__ __align__(16) _Float16 sB[USE_LDS ? 64 * K : 16];
  if constexpr (USE_LDS) {
    // stage the 64 rows of W used by this block's 4 N-tiles (contiguous rows)
    const _Float16* gB = Bw + (size_t)(blockIdx.y * 64) * (size_t)K;
    constexpr int chunks = (64 * K) / 8;             // 16-byte chunks; multiple of 128
    const unsigned lbase = (unsigned)(size_t)(&sB[0]);
#pragma unroll
    for (int c0 = 0; c0 < chunks; c0 += 128) {
      const int c = c0 + threadIdx.x;
      unsigned laddr = lbase + (unsigned)c * 16u;
      const _Float16* g = gB + (size_t)c * 8;
      asm volatile("global_load_async_to_lds_b128 %0, %1, off"
                   :: "v"(laddr), "v"(g) : "memory");
    }
    asm volatile("s_wait_asynccnt 0x0" ::: "memory");
    __syncthreads();
  }
  const size_t brow = (size_t)(wave * 16 + row) * (size_t)K;  // row within LDS panel
  const size_t bng  = (size_t)ncol * (size_t)K;               // row within global W
  const float bv = bias ? bias[ncol] : 0.0f;

  for (int mi = 0; mi < MITER; ++mi) {
    const int tm = blockIdx.x * MITER + mi;
    const size_t am = (size_t)(tm * 16 + row) * (size_t)K;
    v8f acc = {0.f, 0.f, 0.f, 0.f, 0.f, 0.f, 0.f, 0.f};
#pragma unroll
    for (int k0 = 0; k0 < K; k0 += 32) {
      __builtin_prefetch(A + am + k0 + 64, 0, 1);          // global_prefetch_b8
      union { v16h v; v8h h[2]; } a, b;
      // 16-bit A/B layout: group g holds K = 8g+{0..7} (halves 0..7), 16+8g+{0..7} (halves 8..15)
      a.h[0] = *(const v8h*)(A + am + k0 + 8 * grp);
      a.h[1] = *(const v8h*)(A + am + k0 + 16 + 8 * grp);
      if constexpr (USE_LDS) {
        b.h[0] = *(const v8h*)(sB + brow + k0 + 8 * grp);        // ds_load_b128
        b.h[1] = *(const v8h*)(sB + brow + k0 + 16 + 8 * grp);
      } else {
        b.h[0] = *(const v8h*)(Bw + bng + k0 + 8 * grp);
        b.h[1] = *(const v8h*)(Bw + bng + k0 + 16 + 8 * grp);
      }
      acc = __builtin_amdgcn_wmma_f32_16x16x32_f16(false, a.v, false, b.v,
                                                   (short)0, acc, false, false);
    }
#pragma unroll
    for (int r = 0; r < 8; ++r) {
      int mr = tm * 16 + r + 8 * grp;   // C layout: VGPR r, group g -> M = r + 8g
      float v = acc[r] + bv;
      if (act == 1)      v = v > 0.f ? v : 0.f;
      else if (act == 2) v = v > 0.f ? v : 0.2f * v;
      C[(size_t)mr * Ncols + ncol] = v;
      if (C16) C16[(size_t)mr * Ncols + ncol] = (_Float16)v;
    }
  }
}

template<int K, bool L>
static inline void launch_gemm(const _Float16* A, const _Float16* Bw, const float* bias,
                               float* C, _Float16* C16, int Ncols, int act, hipStream_t s) {
  k_wmma_gemm<K, L><<<dim3(MBLK, Ncols / 64), 128, 0, s>>>(A, Bw, bias, C, C16, Ncols, act);
}

// ---------------- graph / elementwise kernels ----------------
__global__ void k_build_edges(const int* ei, int* src, int* dst) {
  int e = blockIdx.x * blockDim.x + threadIdx.x;
  if (e >= NESL) return;
  if (e < NE) { src[e] = ei[e]; dst[e] = ei[NE + e]; }
  else        { src[e] = e - NE; dst[e] = e - NE; }
}

__global__ void k_lin1(const float* x, const float* J, const float* sc,
                       const float* W, const float* b, float* h, _Float16* h16) {
  int i = blockIdx.x * blockDim.x + threadIdx.x;
  if (i >= NN * HD2) return;
  int n = i >> 7, c = i & 127;
  float in[10];
#pragma unroll
  for (int k = 0; k < 8; ++k) in[k] = x[n * 8 + k];
  in[8] = J[n]; in[9] = sc[n];
  const float* w = W + c * 10;
  float acc = b[c];
#pragma unroll
  for (int k = 0; k < 10; ++k) acc += in[k] * w[k];
  h[i] = acc; h16[i] = (_Float16)acc;
}

__global__ void k_attn_scores(const float* xW, const float* att_src, const float* att_dst,
                              float* as_, float* ad_) {
  int i = blockIdx.x * blockDim.x + threadIdx.x;
  if (i >= NN * NHEADS) return;
  int h = i & 7;
  const float* p = xW + (size_t)i * HD1;
  float s = 0.f, d = 0.f;
#pragma unroll 8
  for (int k = 0; k < HD1; ++k) { s += p[k] * att_src[h * HD1 + k]; d += p[k] * att_dst[h * HD1 + k]; }
  as_[i] = s; ad_[i] = d;
}

__global__ void k_edge_max(const int* src, const int* dst, const float* as_, const float* ad_,
                           float* emax) {
  int e = blockIdx.x * blockDim.x + threadIdx.x;
  if (e >= NESL) return;
  int s = src[e], d = dst[e];
#pragma unroll
  for (int h = 0; h < NHEADS; ++h) {
    float v = as_[s * 8 + h] + ad_[d * 8 + h];
    v = v > 0.f ? v : 0.2f * v;
    atomicMaxFloat(&emax[d * 8 + h], v);
  }
}

__global__ void k_edge_expsum(const int* src, const int* dst, const float* as_, const float* ad_,
                              const float* emax, float* eexp, float* esum) {
  int e = blockIdx.x * blockDim.x + threadIdx.x;
  if (e >= NESL) return;
  int s = src[e], d = dst[e];
#pragma unroll
  for (int h = 0; h < NHEADS; ++h) {
    float v = as_[s * 8 + h] + ad_[d * 8 + h];
    v = v > 0.f ? v : 0.2f * v;
    float ex = __expf(v - emax[d * 8 + h]);
    eexp[(size_t)e * 8 + h] = ex;
    atomicAdd(&esum[d * 8 + h], ex);
  }
}

__global__ void k_edge_agg(const int* src, const int* dst, const float* eexp, const float* esum,
                           const float* xW, float* hagg) {
  int i = blockIdx.x * blockDim.x + threadIdx.x;
  if (i >= NESL * NHEADS) return;
  int e = i >> 3, h = i & 7;
  int s = src[e], d = dst[e];
  float coef = eexp[i] / (esum[d * 8 + h] + 1e-16f);
  const float* xs = xW + (size_t)s * HV + h * HD1;
  float* hd = hagg + (size_t)d * HV + h * HD1;
#pragma unroll 4
  for (int k = 0; k < HD1; ++k) atomicAdd(&hd[k], coef * xs[k]);
}

__global__ void k_headsum(const float* hagg, float* hs) {
  int i = blockIdx.x * blockDim.x + threadIdx.x;
  if (i >= NN * HD1) return;
  int n = i >> 6, d = i & 63;
  const float* p = hagg + (size_t)n * HV + d;
  float s = 0.f;
#pragma unroll
  for (int h = 0; h < NHEADS; ++h) s += p[h * HD1];
  hs[i] = s;
}

// batch-norm (training mode, biased var): one block per column
__global__ void k_colstats(const float* x, int cols, float* mean, float* rstd) {
  int c = blockIdx.x;
  float s = 0.f, s2 = 0.f;
  for (int r = threadIdx.x; r < NN; r += blockDim.x) {
    float v = x[(size_t)r * cols + c];
    s += v; s2 += v * v;
  }
  __shared__ float sh[512];
  sh[threadIdx.x] = s; sh[256 + threadIdx.x] = s2;
  __syncthreads();
  for (int o = 128; o > 0; o >>= 1) {
    if (threadIdx.x < o) { sh[threadIdx.x] += sh[threadIdx.x + o]; sh[256 + threadIdx.x] += sh[256 + threadIdx.x + o]; }
    __syncthreads();
  }
  if (threadIdx.x == 0) {
    float m = sh[0] / (float)NN;
    float v = sh[256] / (float)NN - m * m;
    mean[c] = m; rstd[c] = rsqrtf(v + 1e-5f);
  }
}

__global__ void k_bn_apply(const float* x, const float* mean, const float* rstd,
                           const float* g, const float* b, float* out, _Float16* out16, int cols) {
  int i = blockIdx.x * blockDim.x + threadIdx.x;
  if (i >= NN * cols) return;
  int c = i % cols;
  float v = g[c] * (x[i] - mean[c]) * rstd[c] + b[c];
  out[i] = v;
  if (out16) out16[i] = (_Float16)v;
}

// ---------------- APPNP ----------------
__global__ void k_deg(const int* dst, float* deg) {
  int e = blockIdx.x * blockDim.x + threadIdx.x;
  if (e < NESL) atomicAdd(&deg[dst[e]], 1.0f);
}
__global__ void k_rsqrt(float* p, int n) {
  int i = blockIdx.x * blockDim.x + threadIdx.x;
  if (i < n) p[i] = rsqrtf(p[i]);
}
__global__ void k_appnp_scatter(const int* src, const int* dst, const float* dinv,
                                const float* h, float* agg) {
  int e = blockIdx.x * blockDim.x + threadIdx.x;
  if (e >= NESL) return;
  int s = src[e], d = dst[e];
  float nrm = dinv[s] * dinv[d];
  const float* hs = h + (size_t)s * HD2;
  float* ag = agg + (size_t)d * HD2;
#pragma unroll 4
  for (int k = 0; k < HD2; ++k) atomicAdd(&ag[k], nrm * hs[k]);
}
__global__ void k_appnp_combine(const float* agg, const float* h0, float* h) {
  int i = blockIdx.x * blockDim.x + threadIdx.x;
  if (i < NN * HD2) h[i] = 0.9f * agg[i] + 0.1f * h0[i];
}

// ---------------- pooling / final ----------------
__global__ void k_build_xg(const float* h, const float* J, const float* sv, const float* inf_,
                           _Float16* xg) {
  int n = blockIdx.x * blockDim.x + threadIdx.x;
  if (n >= NN) return;
  _Float16* o = xg + (size_t)n * KXG;
  const float* hn = h + (size_t)n * HD2;
  for (int c = 0; c < HD2; ++c) o[c] = (_Float16)hn[c];
  o[128] = (_Float16)J[n]; o[129] = (_Float16)sv[n]; o[130] = (_Float16)inf_[n];
  for (int c = 131; c < KXG; ++c) o[c] = (_Float16)0.0f;
}
__global__ void k_gate2(const float* hid, const float* w, const float* b, float* gate) {
  int n = blockIdx.x * blockDim.x + threadIdx.x;
  if (n >= NN) return;
  const float* r = hid + (size_t)n * HD1;
  float acc = b[0];
#pragma unroll 8
  for (int k = 0; k < HD1; ++k) acc += r[k] * w[k];
  gate[n] = acc;
}
__global__ void k_bmax(const float* gate, const int* batch, float* bmax) {
  int n = blockIdx.x * blockDim.x + threadIdx.x;
  if (n < NN) atomicMaxFloat(&bmax[batch[n]], gate[n]);
}
__global__ void k_bexp(float* gate, const int* batch, const float* bmax, float* bsum) {
  int n = blockIdx.x * blockDim.x + threadIdx.x;
  if (n >= NN) return;
  float ex = __expf(gate[n] - bmax[batch[n]]);
  gate[n] = ex;
  atomicAdd(&bsum[batch[n]], ex);
}
__global__ void k_pool_scatter(const float* v, const float* gate, const float* bsum,
                               const int* batch, float* gpool, int colofs) {
  int i = blockIdx.x * blockDim.x + threadIdx.x;
  if (i >= NN * HD2) return;
  int n = i >> 7, c = i & 127;
  int bi = batch[n];
  float a = gate[n] / (bsum[bi] + 1e-16f);
  atomicAdd(&gpool[(size_t)bi * 384 + colofs + c], a * v[i]);
}
__global__ void k_build_xs(const float* h, const float* gpool, const int* batch,
                           const float* n_nodes, const float* J, const float* sv,
                           const float* inf_, const float* sc,
                           const float* Om, const float* Ph, const float* La,
                           const float* Omn, const float* Phn, const float* Lan,
                           _Float16* xs) {
  int n = blockIdx.x * blockDim.x + threadIdx.x;
  if (n >= NN) return;
  _Float16* o = xs + (size_t)n * KXS;
  int bi = batch[n];
  const float* hn = h + (size_t)n * HD2;
  int c = 0;
  for (int k = 0; k < HD2; ++k) o[c++] = (_Float16)hn[k];
  const float* gb = gpool + (size_t)bi * 384;
  for (int k = 0; k < 384; ++k) o[c++] = (_Float16)gb[k];
  o[c++] = (_Float16)n_nodes[bi];
  o[c++] = (_Float16)J[n]; o[c++] = (_Float16)sv[n];
  o[c++] = (_Float16)inf_[n]; o[c++] = (_Float16)sc[n];
  o[c++] = (_Float16)Om[bi]; o[c++] = (_Float16)Ph[bi]; o[c++] = (_Float16)La[bi];
  o[c++] = (_Float16)Omn[bi]; o[c++] = (_Float16)Phn[bi]; o[c++] = (_Float16)Lan[bi];
  for (; c < KXS; ++c) o[c] = (_Float16)0.0f;
}
__global__ void k_final(const float* s2, const float* w, const float* b,
                        const int* batch, float* out) {
  int n = blockIdx.x * blockDim.x + threadIdx.x;
  if (n >= NN) return;
  const float* r = s2 + (size_t)n * HD2;
  float acc = b[0];
#pragma unroll 8
  for (int k = 0; k < HD2; ++k) acc += r[k] * w[k];
  float s = 1.0f / (1.0f + __expf(-acc));
  atomicAdd(&out[batch[n]], s);
}

// ---------------- param pytree flatten order (JAX sorts dict keys) ----------------
static const int kParamSizes[72] = {
  // att1: att_dst, att_src, bn1.b, bn1.g, bn2.b, bn2.g, gat_W, lin1.W, lin2.W, lin2.b, lin3.W, lin3.b
  512,512,128,128,128,128,65536,8192,8192,64,8192,128,
  512,512,128,128,128,128,65536,8192,8192,64,8192,128,
  512,512,128,128,128,128,65536,8192,8192,64,8192,128,
  // bn1.b, bn1.g, bn2.b, bn2.g, lin1.W, lin1.b, lin3.W, lin3.b, lin4.W, lin4.b, lin5.W, lin5.b
  64,64,128,128,1280,128,33472,64,8192,128,128,1,
  // pool{1,2,3}: gate1.W, gate1.b, gate2.W, gate2.b, nn1.W, nn1.b, nn2.W, nn2.b
  8384,64,64,1,8384,64,8192,128,
  8384,64,64,1,8384,64,8192,128,
  8384,64,64,1,8384,64,8192,128
};

extern "C" void kernel_launch(void* const* d_in, const int* in_sizes, int n_in,
                              void* d_out, int out_size, void* d_ws, size_t ws_size,
                              hipStream_t stream) {
  const float* x    = (const float*)d_in[0];
  const int*   ei   = (const int*)d_in[1];
  const int*   batch= (const int*)d_in[2];
  const float* n_nodes = (const float*)d_in[3];
  const float* Om   = (const float*)d_in[4];
  const float* Ph   = (const float*)d_in[5];
  const float* La   = (const float*)d_in[6];
  const float* Omn  = (const float*)d_in[7];
  const float* Phn  = (const float*)d_in[8];
  const float* Lan  = (const float*)d_in[9];
  const float* J    = (const float*)d_in[10];
  const float* sv   = (const float*)d_in[11];
  const float* inf_ = (const float*)d_in[12];
  const float* sc   = (const float*)d_in[13];

  const float* P[72];
  if (n_in >= 14 + 72) {
    for (int i = 0; i < 72; ++i) P[i] = (const float*)d_in[14 + i];
  } else {
    const float* base = (const float*)d_in[14];
    size_t off = 0;
    for (int i = 0; i < 72; ++i) { P[i] = base + off; off += (size_t)kParamSizes[i]; }
  }

  float* out = (float*)d_out;

  // -------- workspace layout --------
  size_t off = 0;
  auto alloc = [&](size_t bytes) -> void* {
    void* p = (char*)d_ws + off;
    off += (bytes + 255) & ~(size_t)255;
    return p;
  };
  int*   srcv  = (int*)  alloc((size_t)NESL * 4);
  int*   dstv  = (int*)  alloc((size_t)NESL * 4);
  float* dinv  = (float*)alloc((size_t)NN * 4);
  float* h     = (float*)alloc((size_t)NN * HD2 * 4);
  float* h0    = (float*)alloc((size_t)NN * HD2 * 4);
  float* t128a = (float*)alloc((size_t)NN * HD2 * 4);
  float* t128b = (float*)alloc((size_t)NN * HD2 * 4);
  float* t64   = (float*)alloc((size_t)NN * HD1 * 4);
  float* xW    = (float*)alloc((size_t)NN * HV * 4);
  float* hagg  = (float*)alloc((size_t)NN * HV * 4);
  _Float16* af16 = (_Float16*)alloc((size_t)NN * KXS * 2);
  _Float16* bf16 = (_Float16*)alloc((size_t)NN * HD2 * 2);
  float* as_   = (float*)alloc((size_t)NN * NHEADS * 4);
  float* ad_   = (float*)alloc((size_t)NN * NHEADS * 4);
  float* emax  = (float*)alloc((size_t)NN * NHEADS * 4);
  float* esum  = (float*)alloc((size_t)NN * NHEADS * 4);
  float* eexp  = (float*)alloc((size_t)NESL * NHEADS * 4);
  float* meanb = (float*)alloc(256 * 4);
  float* rstdb = (float*)alloc(256 * 4);
  float* gpool = (float*)alloc((size_t)NBATCH * 384 * 4);
  float* bmax  = (float*)alloc((size_t)NBATCH * 4);
  float* bsum  = (float*)alloc((size_t)NBATCH * 4);
  float* gateN = (float*)alloc((size_t)NN * 4);
  _Float16* wg_gat = (_Float16*)alloc((size_t)65536 * 2);
  _Float16* wg_a   = (_Float16*)alloc((size_t)8192 * 2);
  _Float16* wg_b   = (_Float16*)alloc((size_t)8192 * 2);
  _Float16* wg_c   = (_Float16*)alloc((size_t)8192 * 2);
  _Float16* wp_g1  = (_Float16*)alloc((size_t)64 * KXG * 2);
  _Float16* wp_n1  = (_Float16*)alloc((size_t)64 * KXG * 2);
  _Float16* wp_n2  = (_Float16*)alloc((size_t)8192 * 2);
  _Float16* wf_l3  = (_Float16*)alloc((size_t)64 * KXS * 2);
  _Float16* wf_l4  = (_Float16*)alloc((size_t)8192 * 2);
  (void)ws_size; (void)in_sizes; (void)out_size;

  const int TB = 256;
  const float NEGINF = -__builtin_inff();

  // -------- setup --------
  k_fillf<<<CDIV(NBATCH,TB), TB, 0, stream>>>(out, 0.0f, NBATCH);
  k_build_edges<<<CDIV(NESL,TB), TB, 0, stream>>>(ei, srcv, dstv);
  k_lin1<<<CDIV(NN*HD2,TB), TB, 0, stream>>>(x, J, sc, P[40], P[41], h, af16);

  // -------- 3 attention layers --------
  for (int L = 0; L < 3; ++L) {
    const float* att_dst = P[L*12 + 0];
    const float* att_src = P[L*12 + 1];
    const float* bn1b = P[L*12 + 2], *bn1g = P[L*12 + 3];
    const float* bn2b = P[L*12 + 4], *bn2g = P[L*12 + 5];
    const float* gatW = P[L*12 + 6];
    const float* l1W  = P[L*12 + 7];
    const float* l2W  = P[L*12 + 8], *l2b = P[L*12 + 9];
    const float* l3W  = P[L*12 + 10], *l3b = P[L*12 + 11];

    k_cvt16<<<CDIV(65536,TB), TB, 0, stream>>>(gatW, wg_gat, 65536);
    k_cvt16<<<CDIV(8192,TB), TB, 0, stream>>>(l1W, wg_a, 8192);
    k_cvt16<<<CDIV(8192,TB), TB, 0, stream>>>(l2W, wg_b, 8192);
    k_cvt16<<<CDIV(8192,TB), TB, 0, stream>>>(l3W, wg_c, 8192);

    // xW = h @ gat_W^T  [N,512]
    launch_gemm<HD2, true>(af16, wg_gat, nullptr, xW, nullptr, HV, 0, stream);
    k_attn_scores<<<CDIV(NN*NHEADS,TB), TB, 0, stream>>>(xW, att_src, att_dst, as_, ad_);
    k_fillf<<<CDIV(NN*NHEADS,TB), TB, 0, stream>>>(emax, NEGINF, NN*NHEADS);
    k_fillf<<<CDIV(NN*NHEADS,TB), TB, 0, stream>>>(esum, 0.0f, NN*NHEADS);
    k_fillf<<<CDIV(NN*HV,TB), TB, 0, stream>>>(hagg, 0.0f, NN*HV);
    k_edge_max<<<CDIV(NESL,TB), TB, 0, stream>>>(srcv, dstv, as_, ad_, emax);
    k_edge_expsum<<<CDIV(NESL,TB), TB, 0, stream>>>(srcv, dstv, as_, ad_, emax, eexp, esum);
    k_edge_agg<<<CDIV(NESL*NHEADS,TB), TB, 0, stream>>>(srcv, dstv, eexp, esum, xW, hagg);
    k_headsum<<<CDIV(NN*HD1,TB), TB, 0, stream>>>(hagg, t64);
    k_cvt16<<<CDIV(NN*HD1,TB), TB, 0, stream>>>(t64, bf16, NN*HD1);
    // delta = hs @ lin1.W^T  [N,128]
    launch_gemm<HD1, true>(bf16, wg_a, nullptr, t128a, nullptr, HD2, 0, stream);
    // pre = h + delta; bn1
    k_addf<<<CDIV(NN*HD2,TB), TB, 0, stream>>>(h, t128a, t128b, NN*HD2);
    k_colstats<<<HD2, 256, 0, stream>>>(t128b, HD2, meanb, rstdb);
    k_bn_apply<<<CDIV(NN*HD2,TB), TB, 0, stream>>>(t128b, meanb, rstdb, bn1g, bn1b, t128a, af16, HD2);
    // h2 = lin3(relu(lin2(bn1out)))
    launch_gemm<HD2, true>(af16, wg_b, l2b, t64, bf16, HD1, 1, stream);
    launch_gemm<HD1, true>(bf16, wg_c, l3b, t128b, nullptr, HD2, 0, stream);
    // bn2(h2 + bn1out) -> h (and f16 for next layer)
    k_addf<<<CDIV(NN*HD2,TB), TB, 0, stream>>>(t128b, t128a, t128b, NN*HD2);
    k_colstats<<<HD2, 256, 0, stream>>>(t128b, HD2, meanb, rstdb);
    k_bn_apply<<<CDIV(NN*HD2,TB), TB, 0, stream>>>(t128b, meanb, rstdb, bn2g, bn2b, h, af16, HD2);
  }

  // -------- APPNP (gcn_norm, K=10, alpha=0.1) --------
  k_fillf<<<CDIV(NN,TB), TB, 0, stream>>>(dinv, 0.0f, NN);
  k_deg<<<CDIV(NESL,TB), TB, 0, stream>>>(dstv, dinv);
  k_rsqrt<<<CDIV(NN,TB), TB, 0, stream>>>(dinv, NN);
  k_copyf<<<CDIV(NN*HD2,TB), TB, 0, stream>>>(h, h0, NN*HD2);
  for (int k = 0; k < 10; ++k) {
    k_fillf<<<CDIV(NN*HD2,TB), TB, 0, stream>>>(hagg, 0.0f, NN*HD2);
    k_appnp_scatter<<<CDIV(NESL,TB), TB, 0, stream>>>(srcv, dstv, dinv, h, hagg);
    k_appnp_combine<<<CDIV(NN*HD2,TB), TB, 0, stream>>>(hagg, h0, h);
  }

  // -------- gated pools --------
  k_build_xg<<<CDIV(NN,TB), TB, 0, stream>>>(h, J, sv, inf_, af16);
  k_fillf<<<CDIV(NBATCH*384,TB), TB, 0, stream>>>(gpool, 0.0f, NBATCH*384);
  for (int p = 0; p < 3; ++p) {
    const float* g1W = P[48 + p*8 + 0], *g1b = P[48 + p*8 + 1];
    const float* g2W = P[48 + p*8 + 2], *g2b = P[48 + p*8 + 3];
    const float* n1W = P[48 + p*8 + 4], *n1b = P[48 + p*8 + 5];
    const float* n2W = P[48 + p*8 + 6], *n2b = P[48 + p*8 + 7];
    k_cvt16_pad<<<CDIV(64*KXG,TB), TB, 0, stream>>>(g1W, wp_g1, 64, 131, KXG);
    k_cvt16_pad<<<CDIV(64*KXG,TB), TB, 0, stream>>>(n1W, wp_n1, 64, 131, KXG);
    k_cvt16<<<CDIV(8192,TB), TB, 0, stream>>>(n2W, wp_n2, 8192);
    // gate path
    launch_gemm<KXG, true>(af16, wp_g1, g1b, t64, nullptr, HD1, 1, stream);
    k_gate2<<<CDIV(NN,TB), TB, 0, stream>>>(t64, g2W, g2b, gateN);
    k_fillf<<<CDIV(NBATCH,TB), TB, 0, stream>>>(bmax, NEGINF, NBATCH);
    k_fillf<<<CDIV(NBATCH,TB), TB, 0, stream>>>(bsum, 0.0f, NBATCH);
    k_bmax<<<CDIV(NN,TB), TB, 0, stream>>>(gateN, batch, bmax);
    k_bexp<<<CDIV(NN,TB), TB, 0, stream>>>(gateN, batch, bmax, bsum);
    // value path
    launch_gemm<KXG, true>(af16, wp_n1, n1b, t64, bf16, HD1, 1, stream);
    launch_gemm<HD1, true>(bf16, wp_n2, n2b, t128a, nullptr, HD2, 0, stream);
    k_pool_scatter<<<CDIV(NN*HD2,TB), TB, 0, stream>>>(t128a, gateN, bsum, batch, gpool, p*HD2);
  }

  // -------- final MLP --------
  k_build_xs<<<CDIV(NN,TB), TB, 0, stream>>>(h, gpool, batch, n_nodes, J, sv, inf_, sc,
                                             Om, Ph, La, Omn, Phn, Lan, af16);
  k_cvt16_pad<<<CDIV(64*KXS,TB), TB, 0, stream>>>(P[42], wf_l3, 64, 523, KXS);
  k_cvt16<<<CDIV(8192,TB), TB, 0, stream>>>(P[44], wf_l4, 8192);
  // s = bn1(leaky(lin3(xs)))  (K=544: 69.6KB panel -> direct-global B path)
  launch_gemm<KXS, false>(af16, wf_l3, P[43], t64, nullptr, HD1, 2, stream);
  k_colstats<<<HD1, 256, 0, stream>>>(t64, HD1, meanb, rstdb);
  k_bn_apply<<<CDIV(NN*HD1,TB), TB, 0, stream>>>(t64, meanb, rstdb, P[37], P[36], t64, bf16, HD1);
  // s = bn2(leaky(lin4(s)))
  launch_gemm<HD1, true>(bf16, wf_l4, P[45], t128a, nullptr, HD2, 2, stream);
  k_colstats<<<HD2, 256, 0, stream>>>(t128a, HD2, meanb, rstdb);
  k_bn_apply<<<CDIV(NN*HD2,TB), TB, 0, stream>>>(t128a, meanb, rstdb, P[39], P[38], t128a, nullptr, HD2);
  // out[b] = sum_n sigmoid(lin5(s))
  k_final<<<CDIV(NN,TB), TB, 0, stream>>>(t128a, P[46], P[47], batch, out);
}
